// KNNReader_335007450001
// MI455X (gfx1250) — matrix operand order repaired and checked by reading, not compiled, not run on hardware
//
#include <hip/hip_runtime.h>
#include <hip/hip_bf16.h>

// ---------------------------------------------------------------------------
// KNN classifier for MI455X (gfx1250, wave32, WMMA)
//   x[2048,512] f32, keys[65536,512] f32, vals[65536] i32 -> out[2048]
// bf16 WMMA GEMM (keys-bf16 = 64MB, L2-resident), squared-distance ranking,
// threshold-filtered top-10, 8-way N-split + finalize merge for occupancy.
// ---------------------------------------------------------------------------

typedef __attribute__((ext_vector_type(16))) __bf16 v16bf;
typedef __attribute__((ext_vector_type(8)))  float  v8f;
typedef __attribute__((ext_vector_type(4)))  unsigned int u32x4;

#define KDIM    512
#define NQ      2048
#define NKEYS   65536
#define MQ      32                  // queries per workgroup (2 WMMA M-tiles)
#define NCHUNK  128                 // keys per iteration (8 waves x 16)
#define NSPLIT  8                   // key-dimension split across blocks
#define NSLICE  (NKEYS / NSPLIT)    // 8192 keys per block
#define QCAP    2048                // queue cap = 16 rows x 128 cols per phase
#define TOPK    10
#define BIGF    3.4028235e38f

__device__ __forceinline__ unsigned short f2bf_bits(float f) {
    unsigned int u = __builtin_bit_cast(unsigned int, f);
    unsigned int r = u + 0x7FFFu + ((u >> 16) & 1u);
    return (unsigned short)(r >> 16);
}

// One wave per row: convert 512 f32 -> bf16 and compute sum of squares.
__global__ __launch_bounds__(256) void cvt_rows_kernel(
    const float* __restrict__ src, __bf16* __restrict__ dst,
    float* __restrict__ norms, int nrows)
{
    int wave = threadIdx.x >> 5;
    int lane = threadIdx.x & 31;
    int row  = blockIdx.x * 8 + wave;
    if (row >= nrows) return;

    const float4* s = (const float4*)(src + (size_t)row * KDIM + lane * 16);
    unsigned int packed[8];
    float acc = 0.0f;
#pragma unroll
    for (int i = 0; i < 4; ++i) {
        float4 f = s[i];
        acc += f.x * f.x + f.y * f.y + f.z * f.z + f.w * f.w;
        packed[i * 2 + 0] = (unsigned int)f2bf_bits(f.x) | ((unsigned int)f2bf_bits(f.y) << 16);
        packed[i * 2 + 1] = (unsigned int)f2bf_bits(f.z) | ((unsigned int)f2bf_bits(f.w) << 16);
    }
    u32x4* d = (u32x4*)(dst + (size_t)row * KDIM + lane * 16);
    d[0] = (u32x4){packed[0], packed[1], packed[2], packed[3]};
    d[1] = (u32x4){packed[4], packed[5], packed[6], packed[7]};

#pragma unroll
    for (int off = 16; off > 0; off >>= 1)
        acc += __shfl_xor(acc, off, 32);
    if (lane == 0) norms[row] = acc;
}

// GEMM + partial top-10 over one key slice. Block = 32 queries x 8192 keys,
// 8 waves; each wave: 16 keys/chunk, 2 WMMAs per B fragment (B reuse).
__global__ __launch_bounds__(256) void knn_partial_kernel(
    const __bf16* __restrict__ xb,  const float* __restrict__ xnorm,
    const __bf16* __restrict__ kb,  const float* __restrict__ knorm,
    float* __restrict__ partD, int* __restrict__ partI)
{
    __shared__ __align__(16) __bf16 As[MQ * KDIM];   // 32 KB, reused for all keys
    __shared__ float sXn[MQ];
    __shared__ float sThr[MQ];
    __shared__ float sTopD[MQ][TOPK];
    __shared__ int   sTopI[MQ][TOPK];
    __shared__ float qD[QCAP];                       // 8 KB
    __shared__ int   qM[QCAP];                       // 8 KB : (key << 5) | query
    __shared__ int   qCnt;

    const int tid    = threadIdx.x;
    const int slice  = blockIdx.x;
    const int qBase  = blockIdx.y * MQ;
    const int nStart = slice * NSLICE;

    // Stage A tile: 32x512 bf16 = 2048 x 16B
    {
        const u32x4* s = (const u32x4*)(xb + (size_t)qBase * KDIM);
        u32x4*       d = (u32x4*)As;
        for (int i = tid; i < MQ * KDIM / 8; i += 256) d[i] = s[i];
    }
    if (tid < MQ) {
        sXn[tid]  = xnorm[qBase + tid];
        sThr[tid] = BIGF;
#pragma unroll
        for (int j = 0; j < TOPK; ++j) { sTopD[tid][j] = BIGF; sTopI[tid][j] = 0; }
    }
    if (tid == 0) qCnt = 0;
    __syncthreads();

    const int wave = tid >> 5;
    const int lane = tid & 31;
    const int half = lane >> 4;
    const int l16  = lane & 15;

    // A layout (16-bit, interleaved): chunk0 @ k0 + half*8, chunk1 @ k0+16+half*8
    const __bf16* aRow0 = As + l16 * KDIM;             // M-tile 0 (rows 0-15)
    const __bf16* aRow1 = As + (16 + l16) * KDIM;      // M-tile 1 (rows 16-31)

    for (int nOff = 0; nOff < NSLICE; nOff += NCHUNK) {
        const int ng = nStart + nOff + wave * 16 + l16;      // this lane's key
        const __bf16* bRow = kb + (size_t)ng * KDIM + half * 8;

        v8f acc0 = {}, acc1 = {};
#pragma unroll
        for (int ks = 0; ks < KDIM; ks += 32) {
            union { u32x4 q[2]; v16bf v; } bu, a0, a1;
            bu.q[0] = ((const u32x4*)(bRow + ks))[0];
            bu.q[1] = *(const u32x4*)(bRow + ks + 16);
            a0.q[0] = *(const u32x4*)(aRow0 + ks + half * 8);
            a0.q[1] = *(const u32x4*)(aRow0 + ks + 16 + half * 8);
            a1.q[0] = *(const u32x4*)(aRow1 + ks + half * 8);
            a1.q[1] = *(const u32x4*)(aRow1 + ks + 16 + half * 8);
            acc0 = __builtin_amdgcn_wmma_f32_16x16x32_bf16(
                false, a0.v, false, bu.v, (short)0, acc0, false, false);
            acc1 = __builtin_amdgcn_wmma_f32_16x16x32_bf16(
                false, a1.v, false, bu.v, (short)0, acc1, false, false);
        }

        const float kn = knorm[ng];
        // Two push/merge phases so queue pressure per phase is exactly <= QCAP.
#pragma unroll
        for (int ph = 0; ph < 2; ++ph) {
            const v8f acc = ph ? acc1 : acc0;
#pragma unroll
            for (int r = 0; r < 8; ++r) {
                const int m = ph * 16 + half * 8 + r;        // C: lanes16-31 hold M=r+8
                const float d2 = sXn[m] + kn - 2.0f * acc[r];
                if (d2 < sThr[m]) {
                    int slot = atomicAdd(&qCnt, 1);          // ds_add_rtn_u32
                    if (slot < QCAP) { qD[slot] = d2; qM[slot] = (ng << 5) | m; }
                }
            }
            __syncthreads();
            if (tid < MQ) {
                int cnt = qCnt; if (cnt > QCAP) cnt = QCAP;
                for (int i = 0; i < cnt; ++i) {
                    const int meta = qM[i];
                    if ((meta & 31) == tid) {
                        const float d = qD[i];
                        if (d < sTopD[tid][TOPK - 1]) {
                            const int n = meta >> 5;
                            int j = TOPK - 1;
                            while (j > 0 && sTopD[tid][j - 1] > d) {
                                sTopD[tid][j] = sTopD[tid][j - 1];
                                sTopI[tid][j] = sTopI[tid][j - 1];
                                --j;
                            }
                            sTopD[tid][j] = d; sTopI[tid][j] = n;
                        }
                    }
                }
                sThr[tid] = sTopD[tid][TOPK - 1];   // thresholds only shrink
            }
            __syncthreads();
            if (tid == 0) qCnt = 0;
            __syncthreads();
        }
    }

    // Emit partial top-10 lists for this slice.
    if (tid < MQ) {
        float* pd = partD + ((size_t)(qBase + tid) * NSPLIT + slice) * TOPK;
        int*   pi = partI + ((size_t)(qBase + tid) * NSPLIT + slice) * TOPK;
#pragma unroll
        for (int j = 0; j < TOPK; ++j) { pd[j] = sTopD[tid][j]; pi[j] = sTopI[tid][j]; }
    }
}

// Merge NSPLIT partial lists per query -> global top-10 -> mode (smallest class
// on ties, matching argmax over one-hot counts).
__global__ __launch_bounds__(256) void knn_finalize_kernel(
    const float* __restrict__ partD, const int* __restrict__ partI,
    const int* __restrict__ vals, float* __restrict__ out)
{
    const int q = blockIdx.x * 256 + threadIdx.x;
    if (q >= NQ) return;

    float td[TOPK]; int ti[TOPK];
#pragma unroll
    for (int j = 0; j < TOPK; ++j) { td[j] = BIGF; ti[j] = 0; }

    const float* pd = partD + (size_t)q * NSPLIT * TOPK;
    const int*   pi = partI + (size_t)q * NSPLIT * TOPK;
    for (int i = 0; i < NSPLIT * TOPK; ++i) {
        const float d = pd[i];
        if (d < td[TOPK - 1]) {
            const int n = pi[i];
            int j = TOPK - 1;
            while (j > 0 && td[j - 1] > d) {
                td[j] = td[j - 1]; ti[j] = ti[j - 1]; --j;
            }
            td[j] = d; ti[j] = n;
        }
    }

    int cls[TOPK];
#pragma unroll
    for (int j = 0; j < TOPK; ++j) cls[j] = vals[ti[j]];
    int bestC = 0x7fffffff, bestCnt = 0;
#pragma unroll
    for (int j = 0; j < TOPK; ++j) {
        const int c = cls[j];
        int cc = 0;
#pragma unroll
        for (int k = 0; k < TOPK; ++k) cc += (cls[k] == c) ? 1 : 0;
        if (cc > bestCnt || (cc == bestCnt && c < bestC)) { bestCnt = cc; bestC = c; }
    }
    out[q] = (float)bestC;
}

extern "C" void kernel_launch(void* const* d_in, const int* in_sizes, int n_in,
                              void* d_out, int out_size, void* d_ws, size_t ws_size,
                              hipStream_t stream) {
    const float* x    = (const float*)d_in[0];   // [2048,512]
    /* d_in[1] = ver (scalar, unused) */
    const float* keys = (const float*)d_in[2];   // [65536,512]
    const int*   vals = (const int*)  d_in[3];   // [65536]
    float* out = (float*)d_out;

    // Workspace (~67.6 MB): keys_bf16 | x_bf16 | knorm | xnorm | partD | partI
    char* ws = (char*)d_ws;
    __bf16* keysB = (__bf16*)ws;  ws += (size_t)NKEYS * KDIM * 2;
    __bf16* xB    = (__bf16*)ws;  ws += (size_t)NQ    * KDIM * 2;
    float*  kn    = (float*)ws;   ws += (size_t)NKEYS * 4;
    float*  xn    = (float*)ws;   ws += (size_t)NQ    * 4;
    float*  partD = (float*)ws;   ws += (size_t)NQ * NSPLIT * TOPK * 4;
    int*    partI = (int*)ws;

    cvt_rows_kernel<<<NQ / 8,    256, 0, stream>>>(x,    xB,    xn, NQ);
    cvt_rows_kernel<<<NKEYS / 8, 256, 0, stream>>>(keys, keysB, kn, NKEYS);

    dim3 grid(NSPLIT, NQ / MQ);   // 8 x 64 = 512 workgroups
    knn_partial_kernel<<<grid, 256, 0, stream>>>(xB, xn, keysB, kn, partD, partI);
    knn_finalize_kernel<<<NQ / 256, 256, 0, stream>>>(partD, partI, vals, out);
}
